// MultiHeadAttention_43748536877488
// MI455X (gfx1250) — matrix operand order repaired and checked by reading, not compiled, not run on hardware
//
#include <hip/hip_runtime.h>
#include <hip/hip_bf16.h>

typedef __attribute__((ext_vector_type(16))) _Float16 v16h;
typedef __attribute__((ext_vector_type(8)))  _Float16 v8h;
typedef __attribute__((ext_vector_type(8)))  float    v8f;

#define BATCH 4
#define TQ 2048
#define TK 2048
#define CDIM 1024
#define NHEAD 16
#define DHEAD 64
#define MROWS (BATCH * TQ)   // 8192

// ---------------------------------------------------------------------------
// CDNA5 async global->LDS copy (ASYNCcnt path), 16 bytes per lane.
// ---------------------------------------------------------------------------
typedef __attribute__((address_space(3))) _Float16 lds_h;

__device__ inline void mha_async_b128(const _Float16* g, _Float16* l) {
    unsigned loff = (unsigned)(unsigned long long)(lds_h*)l;   // 32-bit LDS offset
    unsigned long long ga = (unsigned long long)g;
    asm volatile("global_load_async_to_lds_b128 %0, %1, off"
                 :: "v"(loff), "v"(ga) : "memory");
}
__device__ inline void mha_wait_async0() { asm volatile("s_wait_asynccnt 0" ::: "memory"); }
__device__ inline void mha_wait_async4() { asm volatile("s_wait_asynccnt 4" ::: "memory"); }

__device__ inline v8h mha_ld8(const _Float16* p) { return *(const v8h*)p; }
__device__ inline v16h mha_cat8(v8h lo, v8h hi) {
    return __builtin_shufflevector(lo, hi, 0,1,2,3,4,5,6,7,8,9,10,11,12,13,14,15);
}

// ---------------------------------------------------------------------------
// Convert kernels
// ---------------------------------------------------------------------------
__global__ __launch_bounds__(256) void mha_cvt_f16(const float* __restrict__ in,
                                                   _Float16* __restrict__ out, int n) {
    int i = blockIdx.x * 256 + threadIdx.x;
    if (i < n) out[i] = (_Float16)in[i];
}

// in: [K=1024][N=1024] row-major  ->  out: [N][K] (transposed) f16
__global__ __launch_bounds__(256) void mha_cvt_f16_t(const float* __restrict__ in,
                                                     _Float16* __restrict__ out) {
    int i = blockIdx.x * 256 + threadIdx.x;   // over 1024*1024
    int k = i >> 10, n = i & 1023;
    out[((size_t)n << 10) + k] = (_Float16)in[i];
}

__global__ __launch_bounds__(256) void mha_cvt_mask(const int* __restrict__ in,
                                                    float* __restrict__ out, int n) {
    int i = blockIdx.x * 256 + threadIdx.x;
    if (i < n) out[i] = in[i] ? 0.0f : -1e10f;
}

// ---------------------------------------------------------------------------
// Projection GEMM: out[M=8192, N=1024] = A[M,1024] @ W[1024,1024] + bias
// A f16 row-major, Bt = W^T ([n][k]) f16.  Tiles staged via async->LDS.
// MODE 0: f16 out head-split [B,H,T,64]; MODE 2: f16 out [B,H,64,T] (V^T);
// MODE 3: f32 out row-major.
// Block 256 = 8 waves; block tile 128x64; wave tile 32x32 (4 WMMA / k-step).
// ---------------------------------------------------------------------------
template<int MODE>
__global__ __launch_bounds__(256)
void mha_gemm_kernel(const _Float16* __restrict__ A, const _Float16* __restrict__ Bt,
                     const float* __restrict__ bias, _Float16* __restrict__ outH,
                     float* __restrict__ outF)
{
    __shared__ __align__(16) _Float16 lA[128 * 32];
    __shared__ __align__(16) _Float16 lB[64 * 32];

    const int tid  = threadIdx.x;
    const int lane = tid & 31;
    const int wave = tid >> 5;
    const int hl   = lane >> 4;          // lane half (0/1)
    const int l16  = lane & 15;
    const int m0   = blockIdx.y * 128;
    const int n0   = blockIdx.x * 64;
    const int wm   = (wave >> 1) * 32;   // 0,32,64,96
    const int wn   = (wave & 1) * 32;    // 0,32

    v8f acc[2][2];
    for (int i = 0; i < 2; i++) for (int j = 0; j < 2; j++) acc[i][j] = v8f{};

    for (int kk = 0; kk < 1024; kk += 32) {
        // async-stage A tile 128x32 (16 halves/thread) and B tile 64x32 (8/thread)
        {
            int h0 = tid * 16;
            int r = h0 >> 5, c = h0 & 31;
            const _Float16* src = A + (size_t)(m0 + r) * 1024 + kk + c;
            mha_async_b128(src,     &lA[r * 32 + c]);
            mha_async_b128(src + 8, &lA[r * 32 + c + 8]);
        }
        {
            int h0 = tid * 8;
            int r = h0 >> 5, c = h0 & 31;
            mha_async_b128(Bt + (size_t)(n0 + r) * 1024 + kk + c, &lB[r * 32 + c]);
        }
        mha_wait_async0();
        __syncthreads();

        v16h af[2], bf[2];
        for (int i = 0; i < 2; i++) {
            // A frag (ISA layout): elem e -> k = e + 8*(e>=8) + 8*hl
            const _Float16* pa = &lA[(wm + i * 16 + l16) * 32];
            af[i] = mha_cat8(mha_ld8(pa + hl * 8), mha_ld8(pa + 16 + hl * 8));
        }
        for (int j = 0; j < 2; j++) {
            // B frag: elem e -> k = e + 8*(e>=8) + 8*hl ; column n = l16
            const _Float16* pb = &lB[(wn + j * 16 + l16) * 32];
            bf[j] = mha_cat8(mha_ld8(pb + hl * 8), mha_ld8(pb + 16 + hl * 8));
        }
        for (int i = 0; i < 2; i++)
            for (int j = 0; j < 2; j++)
                acc[i][j] = __builtin_amdgcn_wmma_f32_16x16x32_f16(
                    false, af[i], false, bf[j], (short)0, acc[i][j], false, false);
        __syncthreads();
    }

    // epilogue
    for (int i = 0; i < 2; i++)
        for (int j = 0; j < 2; j++) {
            int ncol = n0 + wn + j * 16 + l16;
            float bv = bias[ncol];
            for (int r = 0; r < 8; r++) {
                int m = m0 + wm + i * 16 + r + hl * 8;   // C/D layout: row = r + 8*hl
                float val = acc[i][j][r] + bv;
                int b = m >> 11, t = m & 2047;
                if (MODE == 0) {
                    outH[((((size_t)b * NHEAD + (ncol >> 6)) * 2048 + t) << 6) + (ncol & 63)]
                        = (_Float16)val;
                } else if (MODE == 2) {
                    outH[(((size_t)b * NHEAD + (ncol >> 6)) * 64 + (ncol & 63)) * 2048 + t]
                        = (_Float16)val;
                } else {
                    outF[(size_t)m * 1024 + ncol] = val;
                }
            }
        }
}

// ---------------------------------------------------------------------------
// Flash attention: grid (Tq/64, H, B), block 128 = 4 waves, 16 q-rows/wave.
// K/V slabs are shared by all 4 waves -> double-buffered async->LDS staging.
// S^T = K.Q^T (softmax k-dim in per-lane VGPRs), online softmax,
// O^T = V^T.P^T accumulated in f32.
// ---------------------------------------------------------------------------
__device__ inline void mha_stage_kv(const _Float16* Kp, const _Float16* Vp, int k0,
                                    _Float16* lK, _Float16* lV, int tid) {
    // K slab: 32 rows x 64 halves = 2048 halves; 2 async B128 per thread
    {
        int i0 = tid * 8;
        int r = i0 >> 6, c = i0 & 63;
        mha_async_b128(Kp + (size_t)(k0 + r) * DHEAD + c, lK + i0);
        int i1 = i0 + 1024;
        r = i1 >> 6; c = i1 & 63;
        mha_async_b128(Kp + (size_t)(k0 + r) * DHEAD + c, lK + i1);
    }
    // V^T slab: 64 d-rows x 32 k-halves = 2048 halves; 2 async B128 per thread
    {
        int i0 = tid * 8;
        int r = i0 >> 5, c = i0 & 31;
        mha_async_b128(Vp + (size_t)r * TK + k0 + c, lV + i0);
        int i1 = i0 + 1024;
        r = i1 >> 5; c = i1 & 31;
        mha_async_b128(Vp + (size_t)r * TK + k0 + c, lV + i1);
    }
}

__global__ __launch_bounds__(128)
void mha_attn_kernel(const _Float16* __restrict__ Q, const _Float16* __restrict__ K,
                     const _Float16* __restrict__ VT, const float* __restrict__ MF,
                     _Float16* __restrict__ O)
{
    __shared__ __align__(16) _Float16 lK[2][32 * 64];
    __shared__ __align__(16) _Float16 lV[2][64 * 32];

    const int tid  = threadIdx.x;
    const int lane = tid & 31;
    const int wave = tid >> 5;
    const int hl   = lane >> 4;
    const int l16  = lane & 15;
    const int b    = blockIdx.z, h = blockIdx.y;
    const int q0   = blockIdx.x * 64 + wave * 16;

    const size_t bh = (size_t)b * NHEAD + h;
    const _Float16* Qp = Q + (bh * TQ + q0) * DHEAD;
    const _Float16* Kp = K + bh * TK * DHEAD;
    const _Float16* Vp = VT + bh * DHEAD * TK;
    const float*    Mp = MF + (size_t)b * TK;

    // Q as B-fragments of S^T=K.Q^T: column q=l16, contraction d = e + 8*(e>=8) + 8*hl (+32)
    v16h qb0 = mha_cat8(mha_ld8(Qp + (size_t)l16 * DHEAD + hl * 8), mha_ld8(Qp + (size_t)l16 * DHEAD + 16 + hl * 8));
    v16h qb1 = mha_cat8(mha_ld8(Qp + (size_t)l16 * DHEAD + 32 + hl * 8), mha_ld8(Qp + (size_t)l16 * DHEAD + 48 + hl * 8));

    v8f acc[4];
    for (int t = 0; t < 4; t++) acc[t] = v8f{};
    float mrun = -1e30f, lrun = 0.0f;

    mha_stage_kv(Kp, Vp, 0, &lK[0][0], &lV[0][0], tid);

    for (int k0 = 0; k0 < TK; k0 += 32) {
        const int cur = (k0 >> 5) & 1;
        const bool more = (k0 + 32) < TK;
        if (more) {
            mha_stage_kv(Kp, Vp, k0 + 32, &lK[cur ^ 1][0], &lV[cur ^ 1][0], tid);
            mha_wait_async4();    // release current slab; next slab stays in flight
        } else {
            mha_wait_async0();
        }
        __syncthreads();

        // --- S^T for two 16-row k chunks ---
        v8f st[2];
        for (int c = 0; c < 2; c++) {
            const _Float16* kr = &lK[cur][(c * 16 + l16) * 64];
            v16h ka0 = mha_cat8(mha_ld8(kr + hl * 8),      mha_ld8(kr + 16 + hl * 8));
            v16h ka1 = mha_cat8(mha_ld8(kr + 32 + hl * 8), mha_ld8(kr + 48 + hl * 8));
            v8f z = v8f{};
            z = __builtin_amdgcn_wmma_f32_16x16x32_f16(false, ka0, false, qb0, (short)0, z, false, false);
            z = __builtin_amdgcn_wmma_f32_16x16x32_f16(false, ka1, false, qb1, (short)0, z, false, false);
            st[c] = z;
        }

        // scale 1/sqrt(64) + additive mask; k = k0 + c*16 + r + 8*hl
        v8f mk0 = *(const v8f*)(Mp + k0 + hl * 8);
        v8f mk1 = *(const v8f*)(Mp + k0 + 16 + hl * 8);
        float s0[8], s1[8];
        for (int r = 0; r < 8; r++) {
            s0[r] = st[0][r] * 0.125f + mk0[r];
            s1[r] = st[1][r] * 0.125f + mk1[r];
        }

        // --- online softmax over this 32-k slab (combine lane halves) ---
        float mloc = -1e30f;
        for (int r = 0; r < 8; r++) mloc = fmaxf(mloc, fmaxf(s0[r], s1[r]));
        mloc = fmaxf(mloc, __shfl_xor(mloc, 16, 32));
        float mnew = fmaxf(mrun, mloc);
        float corr = __expf(mrun - mnew);
        float p0[8], p1[8], ls = 0.0f;
        for (int r = 0; r < 8; r++) {
            p0[r] = __expf(s0[r] - mnew);
            p1[r] = __expf(s1[r] - mnew);
            ls += p0[r] + p1[r];
        }
        ls += __shfl_xor(ls, 16, 32);
        lrun = lrun * corr + ls;
        mrun = mnew;
        for (int t = 0; t < 4; t++)
            for (int r = 0; r < 8; r++) acc[t][r] *= corr;

        // --- assemble P^T as WMMA B-fragment (k = e + 16*hl, col q = l16) ---
        v16h pb = v16h{};
        for (int e = 0; e < 8; e++) {
            float o0 = __shfl_xor(p0[e], 16, 32);   // lower lanes get upper half's k=8+e
            float o1 = __shfl_xor(p1[e], 16, 32);   // upper lanes get lower half's k=16+e
            pb[e]     = (_Float16)(hl ? o1    : p0[e]);
            pb[e + 8] = (_Float16)(hl ? p1[e] : o0);
        }

        // --- O^T += V^T . P^T  (4 d-tiles of 16) ---
        for (int t = 0; t < 4; t++) {
            const _Float16* vr = &lV[cur][(t * 16 + l16) * 32];
            v16h va = mha_cat8(mha_ld8(vr + hl * 16), mha_ld8(vr + hl * 16 + 8));
            acc[t] = __builtin_amdgcn_wmma_f32_16x16x32_f16(
                false, va, false, pb, (short)0, acc[t], false, false);
        }
        __syncthreads();
    }

    // epilogue: O[b, q0+l16, h*64 + d] = acc/lrun
    float inv = 1.0f / lrun;
    _Float16* ob = O + ((size_t)b * TQ + q0 + l16) * CDIM + h * DHEAD;
    for (int t = 0; t < 4; t++)
        for (int r = 0; r < 8; r++) {
            int d = t * 16 + r + hl * 8;
            ob[d] = (_Float16)(acc[t][r] * inv);
        }
}

// ---------------------------------------------------------------------------
// Host launch
// ---------------------------------------------------------------------------
extern "C" void kernel_launch(void* const* d_in, const int* in_sizes, int n_in,
                              void* d_out, int out_size, void* d_ws, size_t ws_size,
                              hipStream_t stream) {
    const float* x     = (const float*)d_in[0];
    const float* feats = (const float*)d_in[1];
    const int*   mask  = (const int*)d_in[2];
    const float* wq = (const float*)d_in[3];  const float* bq = (const float*)d_in[4];
    const float* wk = (const float*)d_in[5];  const float* bk = (const float*)d_in[6];
    const float* wv = (const float*)d_in[7];  const float* bv = (const float*)d_in[8];
    const float* wo = (const float*)d_in[9];  const float* bo = (const float*)d_in[10];
    float* out = (float*)d_out;

    const size_t NELT = (size_t)MROWS * CDIM;   // 8,388,608
    const size_t WELT = (size_t)CDIM * CDIM;    // 1,048,576

    char* ws = (char*)d_ws;
    size_t off = 0;
    _Float16* xh  = (_Float16*)(ws + off); off += NELT * 2;
    _Float16* fh  = (_Float16*)(ws + off); off += NELT * 2;
    _Float16* qh  = (_Float16*)(ws + off); off += NELT * 2;
    _Float16* kh  = (_Float16*)(ws + off); off += NELT * 2;
    _Float16* vth = (_Float16*)(ws + off); off += NELT * 2;
    _Float16* ah  = (_Float16*)(ws + off); off += NELT * 2;
    _Float16* wqt = (_Float16*)(ws + off); off += WELT * 2;
    _Float16* wkt = (_Float16*)(ws + off); off += WELT * 2;
    _Float16* wvt = (_Float16*)(ws + off); off += WELT * 2;
    _Float16* wot = (_Float16*)(ws + off); off += WELT * 2;
    float*    mkf = (float*)(ws + off);    off += (size_t)BATCH * TK * 4;

    // converts
    mha_cvt_f16<<<(int)(NELT / 256), 256, 0, stream>>>(x, xh, (int)NELT);
    mha_cvt_f16<<<(int)(NELT / 256), 256, 0, stream>>>(feats, fh, (int)NELT);
    mha_cvt_f16_t<<<(int)(WELT / 256), 256, 0, stream>>>(wq, wqt);
    mha_cvt_f16_t<<<(int)(WELT / 256), 256, 0, stream>>>(wk, wkt);
    mha_cvt_f16_t<<<(int)(WELT / 256), 256, 0, stream>>>(wv, wvt);
    mha_cvt_f16_t<<<(int)(WELT / 256), 256, 0, stream>>>(wo, wot);
    mha_cvt_mask<<<(BATCH * TK) / 256, 256, 0, stream>>>(mask, mkf, BATCH * TK);

    dim3 ggrid(CDIM / 64, MROWS / 128);   // 16 x 64
    // projections
    mha_gemm_kernel<0><<<ggrid, 256, 0, stream>>>(xh, wqt, bq, qh, nullptr);
    mha_gemm_kernel<0><<<ggrid, 256, 0, stream>>>(fh, wkt, bk, kh, nullptr);
    mha_gemm_kernel<2><<<ggrid, 256, 0, stream>>>(fh, wvt, bv, vth, nullptr);

    // fused flash attention
    dim3 agrid(TQ / 64, NHEAD, BATCH);
    mha_attn_kernel<<<agrid, 128, 0, stream>>>(qh, kh, vth, mkf, ah);

    // output projection (f32)
    mha_gemm_kernel<3><<<ggrid, 256, 0, stream>>>(ah, wot, bo, nullptr, out);
}
